// AttentiveReadout_35107062678102
// MI455X (gfx1250) — matrix operand order repaired and checked by reading, not compiled, not run on hardware
//
#include <hip/hip_runtime.h>

typedef __attribute__((ext_vector_type(16))) __bf16 v16bf;
typedef __attribute__((ext_vector_type(8)))  float  v8f;
typedef __attribute__((ext_vector_type(4)))  float  f4;
typedef __attribute__((ext_vector_type(4)))  int    v4i;

#define H   256
#define G   4096
#define NT  16   // N-tiles of 16 cols
#define KS  8    // K-steps of 32
#define PACK_ELEMS (H * H)   // 65536 bf16 = 128 KB

#if __has_builtin(__builtin_amdgcn_global_load_async_to_lds_b128)
#define HAVE_ASYNC_LDS 1
typedef __attribute__((address_space(1))) v4i* gv4ip;   // global (device) int4*
typedef __attribute__((address_space(3))) v4i* lv4ip;   // LDS int4*
#endif

// ---------------------------------------------------------------- init
__global__ void k_init(float* segmax, float* sumexp, float* out, int total) {
    int i = blockIdx.x * blockDim.x + threadIdx.x;
    if (i < G) { segmax[i] = 0.0f; sumexp[i] = 0.0f; }
    if (i < total) out[i] = 0.0f;
}

// ---------------------------------------------------------------- pack W1 -> bf16 B-operand layout
// Layout [ks][nt][lane][16]: lane&15 = column n, lanes 0-15 hold K k0..k0+15,
// lanes 16-31 hold K k0+16..k0+31; 16 contiguous halfs per lane. A ks-slab and
// the whole image are contiguous, so LDS staging is a flat copy.
__global__ void k_pack(const float* __restrict__ W1, __bf16* __restrict__ packW) {
    int tid = blockIdx.x * blockDim.x + threadIdx.x;   // 0 .. 65535
    int e    = tid & 15;
    int lane = (tid >> 4) & 31;
    int nt   = (tid >> 9) & 15;
    int ks   = tid >> 13;
    int n = nt * 16 + (lane & 15);
    int k = ks * 32 + 16 * (lane >> 4) + e;
    packW[tid] = (__bf16)W1[k * H + n];
}

// ---------------------------------------------------------------- scores = tanh(xW1+b1)w2 + seg-max
// Persistent blocks: stage packed W1 (128 KB) into LDS once via async
// global->LDS DMA, then grid-stride over 16-row tiles reading B from LDS.
__global__ void __launch_bounds__(256) k_scores(
        const float* __restrict__ x, const long long* __restrict__ batch,
        const __bf16* __restrict__ packW, const float* __restrict__ b1,
        const float* __restrict__ w2, float* __restrict__ scores,
        unsigned int* __restrict__ segmax, int N, int ntiles)
{
    __shared__ __bf16 sB[PACK_ELEMS];                 // 128 KB of the 320 KB/WGP

    // ---- stage: 65536 halfs / 256 threads = 256 halfs (512 B) per thread
    {
        int t = threadIdx.x;
        #pragma unroll
        for (int c = 0; c < 32; ++c) {
            int elem = (c * 256 + t) * 8;             // 8 halfs = 16 B per chunk
#ifdef HAVE_ASYNC_LDS
            __builtin_amdgcn_global_load_async_to_lds_b128(
                (gv4ip)(packW + elem), (lv4ip)(&sB[elem]), 0, 0);
#else
            *(f4*)(&sB[elem]) = *(const f4*)(packW + elem);
#endif
        }
#ifdef HAVE_ASYNC_LDS
#if __has_builtin(__builtin_amdgcn_s_wait_asynccnt)
        __builtin_amdgcn_s_wait_asynccnt(0);
#else
        asm volatile("s_wait_asynccnt 0x0" ::: "memory");
#endif
#endif
        __syncthreads();
    }

    int wave = threadIdx.x >> 5;
    int lane = threadIdx.x & 31;
    int hi   = lane >> 4;
    int nl   = lane & 15;
    int step = gridDim.x * 8;

    for (int mtile = blockIdx.x * 8 + wave; mtile < ntiles; mtile += step) {
        int row  = mtile * 16 + nl;
        int lrow = row < N ? row : N - 1;             // clamp tail loads, keep EXEC full

        v8f acc[NT];
        #pragma unroll
        for (int t = 0; t < NT; ++t) acc[t] = (v8f){0.f,0.f,0.f,0.f,0.f,0.f,0.f,0.f};

        const float* xrow = x + (long long)lrow * H;
        #pragma unroll 1
        for (int ks = 0; ks < KS; ++ks) {
            int k0 = ks * 32;
            // A layout: elems 0-15 = K[k0+16*hi .. +15]
            f4 a0 = *(const f4*)(xrow + k0 + 16 * hi);
            f4 a1 = *(const f4*)(xrow + k0 + 16 * hi + 4);
            f4 a2 = *(const f4*)(xrow + k0 + 16 * hi + 8);
            f4 a3 = *(const f4*)(xrow + k0 + 16 * hi + 12);
            v16bf A;
            #pragma unroll
            for (int e = 0; e < 4; ++e) {
                A[e]      = (__bf16)a0[e];
                A[4 + e]  = (__bf16)a1[e];
                A[8 + e]  = (__bf16)a2[e];
                A[12 + e] = (__bf16)a3[e];
            }
            #pragma unroll
            for (int nt = 0; nt < NT; ++nt) {
                v16bf B = *(const v16bf*)(&sB[(((ks * NT + nt) * 32) + lane) << 4]);
                acc[nt] = __builtin_amdgcn_wmma_f32_16x16x32_bf16(
                    false, A, false, B, (short)0, acc[nt], false, false);
            }
        }

        // C layout: column n = nt*16 + (lane&15); vgpr r holds row r + 8*hi
        float p[8];
        #pragma unroll
        for (int r = 0; r < 8; ++r) p[r] = 0.0f;
        #pragma unroll
        for (int nt = 0; nt < NT; ++nt) {
            int n = nt * 16 + nl;
            float bb = b1[n];
            float ww = w2[n];
            #pragma unroll
            for (int r = 0; r < 8; ++r)
                p[r] += tanhf(acc[nt][r] + bb) * ww;
        }
        // reduce the 16 column-lanes of each half-wave (xor masks < 16 stay in half)
        #pragma unroll
        for (int r = 0; r < 8; ++r)
            for (int m = 1; m < 16; m <<= 1)
                p[r] += __shfl_xor(p[r], m);

        if (nl == 0) {
            #pragma unroll
            for (int r = 0; r < 8; ++r) {
                int rr = mtile * 16 + 8 * hi + r;
                if (rr < N) {
                    float s = p[r];
                    scores[rr] = s;
                    int g = (int)batch[rr];
                    // scatter-max into zeros == clamp at 0; all vals >= 0 -> uint max == float max
                    atomicMax(&segmax[g], __float_as_uint(fmaxf(s, 0.0f)));
                }
            }
        }
    }
}

// ---------------------------------------------------------------- exp + segment sum (sorted batch)
__global__ void k_exp(const float* __restrict__ scores, const long long* __restrict__ batch,
                      const unsigned int* __restrict__ segmax,
                      float* __restrict__ expsc, float* __restrict__ sumexp, int N)
{
    int i  = blockIdx.x * blockDim.x + threadIdx.x;
    int ii = i < N ? i : N - 1;
    bool valid = i < N;
    int lane = threadIdx.x & 31;

    int   g = (int)batch[ii];
    float m = __uint_as_float(segmax[g]);
    float e = valid ? __expf(scores[ii] - m) : 0.0f;
    if (valid) expsc[ii] = e;

    // segmented inclusive scan over equal-g runs within the wave
    float v = e;
    #pragma unroll
    for (int off = 1; off < 32; off <<= 1) {
        float ov = __shfl_up(v, off);
        int   og = __shfl_up(g, off);
        if (lane >= off && og == g) v += ov;
    }
    int gnext = __shfl_down(g, 1);
    bool leader = (lane == 31) || (gnext != g);
    if (leader && v != 0.0f)
        unsafeAtomicAdd(&sumexp[g], v);
}

// ---------------------------------------------------------------- normalized weights
__global__ void k_norm(const float* __restrict__ expsc, const long long* __restrict__ batch,
                       const float* __restrict__ sumexp, float* __restrict__ normw, int N)
{
    int i = blockIdx.x * blockDim.x + threadIdx.x;
    if (i >= N) return;
    int g = (int)batch[i];
    normw[i] = expsc[i] / sumexp[g];
}

// ---------------------------------------------------------------- weighted segment-sum readout
#define CHUNK 128
__global__ void __launch_bounds__(256) k_readout(
        const float* __restrict__ x, const long long* __restrict__ batch,
        const float* __restrict__ normw, float* __restrict__ out, int N)
{
    int h = threadIdx.x;                          // feature this thread owns
    long long start = (long long)blockIdx.x * CHUNK;
    long long end   = start + CHUNK;
    if (end > N) end = N;
    if (start >= end) return;

    int   gcur = (int)batch[start];
    float acc  = 0.0f;
    for (long long i = start; i < end; ++i) {
        int g = (int)batch[i];
        if (g != gcur) {
            unsafeAtomicAdd(&out[(long long)gcur * H + h], acc);
            acc = 0.0f;
            gcur = g;
        }
        acc += x[i * H + h] * normw[i];
    }
    unsafeAtomicAdd(&out[(long long)gcur * H + h], acc);
}

// ---------------------------------------------------------------- launch
extern "C" void kernel_launch(void* const* d_in, const int* in_sizes, int n_in,
                              void* d_out, int out_size, void* d_ws, size_t ws_size,
                              hipStream_t stream) {
    const float*     x     = (const float*)d_in[0];
    const long long* batch = (const long long*)d_in[1];
    const float*     W1    = (const float*)d_in[2];
    const float*     b1    = (const float*)d_in[3];
    const float*     w2    = (const float*)d_in[4];
    float*           out   = (float*)d_out;

    int N = in_sizes[0] / H;          // 500000

    // workspace carve-out (256 B aligned)
    char* ws = (char*)d_ws;
    size_t off = 0;
    auto carve = [&](size_t bytes) { size_t c = off; off = (off + bytes + 255) & ~(size_t)255; return c; };
    float*        scores = (float*)(ws + carve((size_t)N * 4));       // scores, later reused as normw
    float*        expsc  = (float*)(ws + carve((size_t)N * 4));
    float*        sumexp = (float*)(ws + carve((size_t)G * 4));
    unsigned int* segmax = (unsigned int*)(ws + carve((size_t)G * 4));
    __bf16*       packW  = (__bf16*)(ws + carve((size_t)PACK_ELEMS * 2));

    int ntiles = (N + 15) / 16;       // 31250
    int sblocks = (ntiles + 7) / 8;
    if (sblocks > 1024) sblocks = 1024;   // persistent blocks

    k_init<<<(G * H + 255) / 256, 256, 0, stream>>>((float*)segmax, sumexp, out, G * H);
    k_pack<<<(H * H) / 256, 256, 0, stream>>>(W1, packW);
    k_scores<<<sblocks, 256, 0, stream>>>(x, batch, packW, b1, w2, scores, segmax, N, ntiles);
    k_exp<<<(N + 255) / 256, 256, 0, stream>>>(scores, batch, segmax, expsc, sumexp, N);
    k_norm<<<(N + 255) / 256, 256, 0, stream>>>(expsc, batch, sumexp, scores, N);
    k_readout<<<(N + CHUNK - 1) / CHUNK, 256, 0, stream>>>(x, batch, scores, out, N);
}
